// MoE_85169201479864
// MI455X (gfx1250) — matrix operand order repaired and checked by reading, not compiled, not run on hardware
//
#include <hip/hip_runtime.h>

// ---------------- problem constants ----------------
#define DQ   1024      // embed dim
#define FF   4096      // ffn dim
#define NE   8         // experts
#define CAPE 2560      // capacity = ceil(8192*2*1.25/8), divisible by 256
#define BM   256       // block tile M (8 waves x 2 mtiles x 16 rows)
#define BN   64        // block tile N (4 wmma tiles per wave)
#define BK   32        // K step (bf16 wmma K)
#define LDK  40        // LDS K stride (80B rows: 16B aligned, bank-spread)

typedef __bf16 bf16_t;
typedef bf16_t v16bf __attribute__((ext_vector_type(16)));
typedef bf16_t v8bf  __attribute__((ext_vector_type(8)));
typedef bf16_t v2bf  __attribute__((ext_vector_type(2)));
typedef float  v8f   __attribute__((ext_vector_type(8)));

// ---------------- zero output ----------------
__global__ __launch_bounds__(256) void moe_zero_f4(float4* __restrict__ p, int n4) {
    int i = blockIdx.x * 256 + threadIdx.x;
    if (i < n4) { float4 z; z.x = z.y = z.z = z.w = 0.f; p[i] = z; }
}

// ---------------- router: one wave32 per token ----------------
__global__ __launch_bounds__(256) void moe_router(const float* __restrict__ x,
                                                  const float* __restrict__ Wr,
                                                  int* __restrict__ tope,
                                                  float* __restrict__ topp, int T) {
    int wave = threadIdx.x >> 5, lane = threadIdx.x & 31;
    int t = blockIdx.x * 8 + wave;
    if (t >= T) return;
    const float* xr = x + (size_t)t * DQ;
    float acc[NE];
#pragma unroll
    for (int e = 0; e < NE; ++e) acc[e] = 0.f;
    for (int k = lane; k < DQ; k += 32) {
        float xv = xr[k];
        const float* w = Wr + (size_t)k * NE;   // Wr is (D, E) row-major
#pragma unroll
        for (int e = 0; e < NE; ++e) acc[e] += xv * w[e];
    }
#pragma unroll
    for (int off = 16; off >= 1; off >>= 1)
#pragma unroll
        for (int e = 0; e < NE; ++e) acc[e] += __shfl_xor(acc[e], off, 32);
    if (lane == 0) {
        float m = acc[0];
#pragma unroll
        for (int e = 1; e < NE; ++e) m = fmaxf(m, acc[e]);
        float p[NE], s = 0.f;
#pragma unroll
        for (int e = 0; e < NE; ++e) { p[e] = __expf(acc[e] - m); s += p[e]; }
        float inv = 1.0f / s;
        int b0 = 0;
#pragma unroll
        for (int e = 1; e < NE; ++e) if (p[e] > p[b0]) b0 = e;
        int b1 = (b0 == 0) ? 1 : 0;
#pragma unroll
        for (int e = 0; e < NE; ++e) if (e != b0 && p[e] > p[b1]) b1 = e;
        tope[2 * t + 0] = b0;  topp[2 * t + 0] = p[b0] * inv;
        tope[2 * t + 1] = b1;  topp[2 * t + 1] = p[b1] * inv;
    }
}

// ---------------- deterministic capacity assignment (token-order) ----------------
__global__ __launch_bounds__(1024) void moe_assign(const int* __restrict__ tope,
                                                   const float* __restrict__ topp,
                                                   int* __restrict__ idxb,
                                                   float* __restrict__ gateb, int T) {
    __shared__ int scan[1024];
    __shared__ int sbase;
    int e = blockIdx.x, tid = threadIdx.x;
    if (tid == 0) sbase = 0;
    __syncthreads();
    int nchunk = T / 1024;
    for (int c = 0; c < nchunk; ++c) {
        int t = c * 1024 + tid;
        int e0 = tope[2 * t], e1 = tope[2 * t + 1];
        int flag = (e0 == e) || (e1 == e);
        float g = (e0 == e) ? topp[2 * t] : ((e1 == e) ? topp[2 * t + 1] : 0.0f);
        scan[tid] = flag;
        __syncthreads();
        for (int off = 1; off < 1024; off <<= 1) {
            int v = (tid >= off) ? scan[tid - off] : 0;
            __syncthreads();
            scan[tid] += v;
            __syncthreads();
        }
        int pos = sbase + scan[tid] - 1;   // inclusive scan -> slot
        if (flag && pos < CAPE) { idxb[e * CAPE + pos] = t; gateb[e * CAPE + pos] = g; }
        __syncthreads();
        if (tid == 0) sbase += scan[1023];
        __syncthreads();
    }
    int tot = sbase;
    for (int s = tot + tid; s < CAPE; s += 1024) {   // pad unused slots
        idxb[e * CAPE + s] = 0; gateb[e * CAPE + s] = 0.0f;
    }
}

// ---------------- helpers ----------------
__device__ __forceinline__ v16bf frag_cat(v8bf lo, v8bf hi) {
    return __builtin_shufflevector(lo, hi, 0,1,2,3,4,5,6,7,8,9,10,11,12,13,14,15);
}
__device__ __forceinline__ v8bf pack8(float4 a, float4 b) {
    v8bf r;
    r[0]=(bf16_t)a.x; r[1]=(bf16_t)a.y; r[2]=(bf16_t)a.z; r[3]=(bf16_t)a.w;
    r[4]=(bf16_t)b.x; r[5]=(bf16_t)b.y; r[6]=(bf16_t)b.z; r[7]=(bf16_t)b.w;
    return r;
}

// ---------------- GEMM1: H = relu(gather(x) @ W1[e]), H stored bf16 ----------------
// One-barrier double-buffered pipeline:
//   iter i: issue raw loads(tile i+1) -> WMMA from buf[i%2] -> cvt+store buf[(i+1)%2] -> barrier
__global__ __launch_bounds__(256) void moe_gemm1(const float* __restrict__ x,
                                                 const float* __restrict__ W1e,
                                                 const int* __restrict__ idxe,
                                                 bf16_t* __restrict__ H) {
    __shared__ bf16_t Als[2][BM * LDK];   // [m][k]  (2 x 20480 B)
    __shared__ bf16_t Bls[2][BN * LDK];   // [n][k]  (2 x 5120 B, transposed)
    int tid = threadIdx.x, lane = tid & 31, wave = tid >> 5;
    int n0 = blockIdx.x * BN, m0 = blockIdx.y * BM;

    // --- A staging: 4 chunks of 8 floats (coalesced within each gathered row) ---
    const float* aptr[4]; int ao[4];
#pragma unroll
    for (int i = 0; i < 4; ++i) {
        int c = tid + i * 256;                  // chunk id
        int r = c >> 2, kk = (c & 3) * 8;       // row 0..255, k-offset 0/8/16/24
        int tk = idxe[m0 + r];                  // gathered token (pad slots -> 0)
        aptr[i] = x + (size_t)tk * DQ + kk;
        ao[i]   = r * LDK + kk;
    }
    // --- B staging: 2x4 (k x n) patch, packed to bf16x2, transposed into LDS ---
    int kp = tid >> 4, bn4 = (tid & 15) * 4;    // k-pair 0..15, n 0..60
    const float* bptr = W1e + (size_t)(2 * kp) * FF + n0 + bn4;

    v8f acc[2][4] = {};
    int half = lane >> 4, l15 = lane & 15;
    int afo0 = (wave * 32 + l15) * LDK + half * 8;
    int bkb = half * 8;

    float4 ra0[4], ra1[4], re0, re1;
    // prologue: tile 0 -> regs -> buf 0
#pragma unroll
    for (int i = 0; i < 4; ++i) {
        ra0[i] = *(const float4*)(aptr[i]);
        ra1[i] = *(const float4*)(aptr[i] + 4);
    }
    re0 = *(const float4*)(bptr);
    re1 = *(const float4*)(bptr + FF);
#pragma unroll
    for (int i = 0; i < 4; ++i) *(v8bf*)&Als[0][ao[i]] = pack8(ra0[i], ra1[i]);
    {
        v2bf q0, q1, q2, q3;
        q0[0]=(bf16_t)re0.x; q0[1]=(bf16_t)re1.x;
        q1[0]=(bf16_t)re0.y; q1[1]=(bf16_t)re1.y;
        q2[0]=(bf16_t)re0.z; q2[1]=(bf16_t)re1.z;
        q3[0]=(bf16_t)re0.w; q3[1]=(bf16_t)re1.w;
        *(v2bf*)&Bls[0][(bn4 + 0) * LDK + 2 * kp] = q0;
        *(v2bf*)&Bls[0][(bn4 + 1) * LDK + 2 * kp] = q1;
        *(v2bf*)&Bls[0][(bn4 + 2) * LDK + 2 * kp] = q2;
        *(v2bf*)&Bls[0][(bn4 + 3) * LDK + 2 * kp] = q3;
    }
    __syncthreads();

    for (int k0 = 0; k0 < DQ; k0 += BK) {
        int cur = (k0 >> 5) & 1, nxt = cur ^ 1;
        bool more = (k0 + BK) < DQ;
        if (more) {   // issue raw loads for next tile; no wait until store phase
            int kn = k0 + BK;
#pragma unroll
            for (int i = 0; i < 4; ++i) {
                ra0[i] = *(const float4*)(aptr[i] + kn);
                ra1[i] = *(const float4*)(aptr[i] + kn + 4);
            }
            re0 = *(const float4*)(bptr + (size_t)kn * FF);
            re1 = *(const float4*)(bptr + (size_t)kn * FF + FF);
        }

#pragma unroll
        for (int mt = 0; mt < 2; ++mt) {
            const bf16_t* afp = &Als[cur][afo0 + mt * 16 * LDK];
            v16bf af = frag_cat(*(const v8bf*)(afp), *(const v8bf*)(afp + 16));
#pragma unroll
            for (int j = 0; j < 4; ++j) {
                const bf16_t* bp = &Bls[cur][(j * 16 + l15) * LDK + bkb];
                v16bf bfr = frag_cat(*(const v8bf*)bp, *(const v8bf*)(bp + 16));
                acc[mt][j] = __builtin_amdgcn_wmma_f32_16x16x32_bf16(
                    false, af, false, bfr, (short)0, acc[mt][j], false, false);
            }
        }

        if (more) {   // cvt + publish next tile (loadcnt wait lands here, after WMMA)
#pragma unroll
            for (int i = 0; i < 4; ++i) *(v8bf*)&Als[nxt][ao[i]] = pack8(ra0[i], ra1[i]);
            v2bf q0, q1, q2, q3;
            q0[0]=(bf16_t)re0.x; q0[1]=(bf16_t)re1.x;
            q1[0]=(bf16_t)re0.y; q1[1]=(bf16_t)re1.y;
            q2[0]=(bf16_t)re0.z; q2[1]=(bf16_t)re1.z;
            q3[0]=(bf16_t)re0.w; q3[1]=(bf16_t)re1.w;
            *(v2bf*)&Bls[nxt][(bn4 + 0) * LDK + 2 * kp] = q0;
            *(v2bf*)&Bls[nxt][(bn4 + 1) * LDK + 2 * kp] = q1;
            *(v2bf*)&Bls[nxt][(bn4 + 2) * LDK + 2 * kp] = q2;
            *(v2bf*)&Bls[nxt][(bn4 + 3) * LDK + 2 * kp] = q3;
        }
        __syncthreads();   // single barrier per iteration
    }
    // relu + store bf16 hidden; C/D layout: N=l15, M=vgpr + 8*half
#pragma unroll
    for (int mt = 0; mt < 2; ++mt)
#pragma unroll
        for (int j = 0; j < 4; ++j)
#pragma unroll
            for (int v = 0; v < 8; ++v) {
                int mm = m0 + wave * 32 + mt * 16 + half * 8 + v;
                int nn = n0 + j * 16 + l15;
                H[(size_t)mm * FF + nn] = (bf16_t)fmaxf(acc[mt][j][v], 0.0f);
            }
}

// ---------------- GEMM2: y[idx] += gate * (H @ W2[e]) ----------------
// A tile (bf16, no conversion) is DMA'd with CDNA5 async global->LDS copies into a
// double-buffered LDS region; B tile uses the register path (fp32->bf16 cvt).
// One barrier per iteration: asyncs for buf[nxt] issued at iteration start (readers
// of that buffer finished before the previous barrier); s_wait_asynccnt 0 precedes
// the publishing barrier.
#define ABUF_B (BM * LDK * 2)            // one A buffer in bytes (20480)
__global__ __launch_bounds__(256) void moe_gemm2(const bf16_t* __restrict__ H,
                                                 const float* __restrict__ W2e,
                                                 const int* __restrict__ idxe,
                                                 const float* __restrict__ gatee,
                                                 float* __restrict__ y) {
    extern __shared__ bf16_t smem[];     // [2][BM*LDK] A + [2][BN*LDK] B
    bf16_t* Bbase = smem + 2 * BM * LDK;
    int tid = threadIdx.x, lane = tid & 31, wave = tid >> 5;
    int n0 = blockIdx.x * BN, m0 = blockIdx.y * BM;

    const bf16_t* aptr[4]; unsigned alo[4];
    unsigned abase = (unsigned)(uintptr_t)(void*)smem;   // LDS byte offset of smem
#pragma unroll
    for (int i = 0; i < 4; ++i) {
        int c = tid + i * 256;
        int r = c >> 2, kk = (c & 3) * 8;
        aptr[i] = H + (size_t)(m0 + r) * FF + kk;
        alo[i]  = abase + (unsigned)(r * LDK + kk) * 2;  // buffer-0 LDS byte address
    }
    int kp = tid >> 4, bn4 = (tid & 15) * 4;
    const float* bptr = W2e + (size_t)(2 * kp) * DQ + n0 + bn4;

    v8f acc[2][4] = {};
    int half = lane >> 4, l15 = lane & 15;
    int afo0 = (wave * 32 + l15) * LDK + half * 8;
    int bkb = half * 8;

    // prologue: async A tile 0 -> buffer 0; B tile 0 -> regs -> buffer 0
#pragma unroll
    for (int i = 0; i < 4; ++i) {
        unsigned long long ga = (unsigned long long)(uintptr_t)(aptr[i]);
        asm volatile("global_load_async_to_lds_b128 %0, %1, off"
                     :: "v"(alo[i]), "v"(ga) : "memory");
    }
    float4 re0 = *(const float4*)(bptr);
    float4 re1 = *(const float4*)(bptr + DQ);
    {
        v2bf q0, q1, q2, q3;
        q0[0]=(bf16_t)re0.x; q0[1]=(bf16_t)re1.x;
        q1[0]=(bf16_t)re0.y; q1[1]=(bf16_t)re1.y;
        q2[0]=(bf16_t)re0.z; q2[1]=(bf16_t)re1.z;
        q3[0]=(bf16_t)re0.w; q3[1]=(bf16_t)re1.w;
        *(v2bf*)&Bbase[(bn4 + 0) * LDK + 2 * kp] = q0;
        *(v2bf*)&Bbase[(bn4 + 1) * LDK + 2 * kp] = q1;
        *(v2bf*)&Bbase[(bn4 + 2) * LDK + 2 * kp] = q2;
        *(v2bf*)&Bbase[(bn4 + 3) * LDK + 2 * kp] = q3;
    }
    asm volatile("s_wait_asynccnt 0x0" ::: "memory");
    __syncthreads();

    for (int k0 = 0; k0 < FF; k0 += BK) {
        int cur = (k0 >> 5) & 1, nxt = cur ^ 1;
        bool more = (k0 + BK) < FF;
        if (more) {   // fill A buf[nxt] via async DMA while we read buf[cur]
            int kn = k0 + BK;
#pragma unroll
            for (int i = 0; i < 4; ++i) {
                unsigned long long ga = (unsigned long long)(uintptr_t)(aptr[i] + kn);
                unsigned la = alo[i] + (unsigned)nxt * ABUF_B;
                asm volatile("global_load_async_to_lds_b128 %0, %1, off"
                             :: "v"(la), "v"(ga) : "memory");
            }
            re0 = *(const float4*)(bptr + (size_t)kn * DQ);
            re1 = *(const float4*)(bptr + (size_t)kn * DQ + DQ);
        }

        const bf16_t* Ap = smem + cur * (BM * LDK);
        const bf16_t* Bp = Bbase + cur * (BN * LDK);
#pragma unroll
        for (int mt = 0; mt < 2; ++mt) {
            const bf16_t* afp = Ap + afo0 + mt * 16 * LDK;
            v16bf af = frag_cat(*(const v8bf*)(afp), *(const v8bf*)(afp + 16));
#pragma unroll
            for (int j = 0; j < 4; ++j) {
                const bf16_t* bp = Bp + (j * 16 + l15) * LDK + bkb;
                v16bf bfr = frag_cat(*(const v8bf*)bp, *(const v8bf*)(bp + 16));
                acc[mt][j] = __builtin_amdgcn_wmma_f32_16x16x32_bf16(
                    false, af, false, bfr, (short)0, acc[mt][j], false, false);
            }
        }

        if (more) {   // cvt + publish next B tile (loadcnt wait lands here)
            v2bf q0, q1, q2, q3;
            q0[0]=(bf16_t)re0.x; q0[1]=(bf16_t)re1.x;
            q1[0]=(bf16_t)re0.y; q1[1]=(bf16_t)re1.y;
            q2[0]=(bf16_t)re0.z; q2[1]=(bf16_t)re1.z;
            q3[0]=(bf16_t)re0.w; q3[1]=(bf16_t)re1.w;
            bf16_t* Bn = Bbase + nxt * (BN * LDK);
            *(v2bf*)&Bn[(bn4 + 0) * LDK + 2 * kp] = q0;
            *(v2bf*)&Bn[(bn4 + 1) * LDK + 2 * kp] = q1;
            *(v2bf*)&Bn[(bn4 + 2) * LDK + 2 * kp] = q2;
            *(v2bf*)&Bn[(bn4 + 3) * LDK + 2 * kp] = q3;
        }
        asm volatile("s_wait_asynccnt 0x0" ::: "memory");  // A buf[nxt] complete
        __syncthreads();   // single barrier per iteration
    }
    // scatter-combine; experts serialized + unique slot->token within expert:
    // atomicAdd is contention-free here (deterministic), gate=0 kills pad slots.
#pragma unroll
    for (int mt = 0; mt < 2; ++mt)
#pragma unroll
        for (int v = 0; v < 8; ++v) {
            int slot = m0 + wave * 32 + mt * 16 + half * 8 + v;
            int tok  = idxe[slot];
            float g  = gatee[slot];
            float* yr = y + (size_t)tok * DQ + n0 + l15;
#pragma unroll
            for (int j = 0; j < 4; ++j)
                atomicAdd(yr + j * 16, g * acc[mt][j][v]);
        }
}

// ---------------- launch ----------------
extern "C" void kernel_launch(void* const* d_in, const int* in_sizes, int n_in,
                              void* d_out, int out_size, void* d_ws, size_t ws_size,
                              hipStream_t stream) {
    const float* x  = (const float*)d_in[0];
    const float* Wr = (const float*)d_in[1];
    const float* W1 = (const float*)d_in[2];
    const float* W2 = (const float*)d_in[3];
    float* y = (float*)d_out;
    int T = in_sizes[0] / DQ;   // 8192 tokens

    char* ws = (char*)d_ws;
    int*    idxb  = (int*)ws;                                  // NE*CAPE ints
    float*  gateb = (float*)(ws + (size_t)NE * CAPE * 4);      // NE*CAPE floats
    char*   p2    = ws + (size_t)2 * NE * CAPE * 4;
    int*    tope  = (int*)p2;                                  // T*2 ints
    float*  topp  = (float*)(p2 + (size_t)T * 2 * 4);          // T*2 floats
    size_t  hoff  = ((size_t)2 * NE * CAPE * 4 + (size_t)T * 4 * 4 + 255) & ~(size_t)255;
    bf16_t* H     = (bf16_t*)(ws + hoff);                      // CAPE*FF bf16 (~20MB), reused

    int n4 = out_size / 4;
    moe_zero_f4<<<(n4 + 255) / 256, 256, 0, stream>>>((float4*)y, n4);
    moe_router<<<T / 8, 256, 0, stream>>>(x, Wr, tope, topp, T);
    moe_assign<<<NE, 1024, 0, stream>>>(tope, topp, idxb, gateb, T);
    size_t smem2 = (size_t)(2 * BM * LDK + 2 * BN * LDK) * sizeof(bf16_t);  // 51200 B
    for (int e = 0; e < NE; ++e) {   // serialized per expert -> race-free scatter
        moe_gemm1<<<dim3(FF / BN, CAPE / BM), 256, 0, stream>>>(
            x, W1 + (size_t)e * DQ * FF, idxb + e * CAPE, H);
        moe_gemm2<<<dim3(DQ / BN, CAPE / BM), 256, smem2, stream>>>(
            H, W2 + (size_t)e * FF * DQ, idxb + e * CAPE, gateb + e * CAPE, y);
    }
}
